// Encoder_39797166965032
// MI455X (gfx1250) — matrix operand order repaired and checked by reading, not compiled, not run on hardware
//
#include <hip/hip_runtime.h>
#include <hip/hip_bf16.h>
#include <stdint.h>

#define LN_EPS   1e-5f
#define NNODES   50000
#define NEDGES   800000

typedef _Float16 v16h __attribute__((ext_vector_type(16)));
typedef _Float16 v8h  __attribute__((ext_vector_type(8)));
typedef float    v8f  __attribute__((ext_vector_type(8)));

static __device__ __forceinline__ v16h cat8(v8h lo, v8h hi) {
  return __builtin_shufflevector(lo, hi, 0,1,2,3,4,5,6,7,8,9,10,11,12,13,14,15);
}

// ---------------------------------------------------------------------------
// Wave-level 16-row MLP:  tile[16 x Kpad] (f16, LDS)  @ W1T -> +b1 -> LN ->
// ReLU -> hid[16 x 64] (f16, LDS) @ W2T -> acc2 (f32, C/D layout, no b2).
// Uses v_wmma_f32_16x16x32_f16.  EXEC must be full (wave-uniform callers).
// ---------------------------------------------------------------------------
static __device__ __forceinline__ void wave_mlp(
    const _Float16* tile, int Kpad, _Float16* hid,
    const _Float16* __restrict__ W1T, const float* __restrict__ b1,
    const float* __restrict__ g,     const float* __restrict__ be,
    const _Float16* __restrict__ W2T, int outTiles, int lane, v8f* acc2)
{
  const int half = lane >> 4;
  const int l16  = lane & 15;

  v8f acc[4];
#pragma unroll
  for (int nt = 0; nt < 4; ++nt)
    acc[nt] = (v8f){0.f,0.f,0.f,0.f,0.f,0.f,0.f,0.f};

  const int nk = Kpad >> 5;
  for (int kc = 0; kc < nk; ++kc) {
    // A fragment: row = l16, K = kc*32 + 8*half + {0..7, 16..23}
    const _Float16* ap = tile + l16 * Kpad + (kc << 5) + 8 * half;
    v16h a = cat8(*(const v8h*)ap, *(const v8h*)(ap + 16));
#pragma unroll
    for (int nt = 0; nt < 4; ++nt) {
      // B fragment: col = nt*16+l16, K = kc*32 + 8*half + {0..7, 16..23}
      const _Float16* bp = W1T + (size_t)((nt << 4) + l16) * Kpad + (kc << 5) + 8 * half;
      v16h b = cat8(*(const v8h*)bp, *(const v8h*)(bp + 16));
      acc[nt] = __builtin_amdgcn_wmma_f32_16x16x32_f16(
          false, a, false, b, (short)0, acc[nt], false, false);
    }
  }

  // bias + LayerNorm + ReLU in C/D layout; store hidden to LDS as f16
  float gc[4], bec[4];
#pragma unroll
  for (int nt = 0; nt < 4; ++nt) {
    const int col = (nt << 4) + l16;
    const float bb = b1[col];
    gc[nt] = g[col]; bec[nt] = be[col];
#pragma unroll
    for (int i = 0; i < 8; ++i) acc[nt][i] += bb;
  }
#pragma unroll
  for (int i = 0; i < 8; ++i) {
    float s  = acc[0][i] + acc[1][i] + acc[2][i] + acc[3][i];
    float s2 = acc[0][i]*acc[0][i] + acc[1][i]*acc[1][i]
             + acc[2][i]*acc[2][i] + acc[3][i]*acc[3][i];
#pragma unroll
    for (int m = 1; m < 16; m <<= 1) {   // reduce within 16-lane group
      s  += __shfl_xor(s,  m, 32);
      s2 += __shfl_xor(s2, m, 32);
    }
    const float mean = s * (1.f / 64.f);
    const float var  = s2 * (1.f / 64.f) - mean * mean;
    const float rs   = rsqrtf(var + LN_EPS);
    const int   row  = i + (half << 3);
#pragma unroll
    for (int nt = 0; nt < 4; ++nt) {
      float h = (acc[nt][i] - mean) * rs * gc[nt] + bec[nt];
      h = fmaxf(h, 0.f);
      hid[row * 64 + (nt << 4) + l16] = (_Float16)h;
    }
  }

  // GEMM2: hid[16x64] @ W2T -> acc2 (in-order per-wave DS guarantees visibility)
#pragma unroll
  for (int nt = 0; nt < 4; ++nt)
    acc2[nt] = (v8f){0.f,0.f,0.f,0.f,0.f,0.f,0.f,0.f};
#pragma unroll
  for (int kc = 0; kc < 2; ++kc) {
    const _Float16* ap = hid + l16 * 64 + (kc << 5) + 8 * half;
    v16h a = cat8(*(const v8h*)ap, *(const v8h*)(ap + 16));
    for (int nt = 0; nt < outTiles; ++nt) {
      const _Float16* bp = W2T + (size_t)((nt << 4) + l16) * 64 + (kc << 5) + 8 * half;
      v16h b = cat8(*(const v8h*)bp, *(const v8h*)(bp + 16));
      acc2[nt] = __builtin_amdgcn_wmma_f32_16x16x32_f16(
          false, a, false, b, (short)0, acc2[nt], false, false);
    }
  }
}

// ---------------------------------------------------------------------------
// Edge MLP + scatter-sum.  4 waves / block, one 16-edge tile per wave.
// mode: 0 = store e_new (f16 ebuf), 1 = residual (ebuf += e_new),
//       2 = final layer (e_out width 1 to d_out, agg width 1)
// ---------------------------------------------------------------------------
__global__ __launch_bounds__(128) void edge_mlp_kernel(
    int E, const int* __restrict__ srcIdx, const int* __restrict__ dstIdx,
    const float* __restrict__ nodeSrc, int nin,
    const float* __restrict__ eattrF32, _Float16* ebuf, int ein, int eIsF16,
    int Kpad,
    const _Float16* __restrict__ W1T, const float* __restrict__ b1,
    const float* __restrict__ g,     const float* __restrict__ be,
    const _Float16* __restrict__ W2T, const float* __restrict__ b2,
    int outTiles, int outValid, int mode,
    float* __restrict__ edgeOutF32, float* agg)
{
  __shared__ _Float16 sTile[4][16 * 192];
  __shared__ _Float16 sHid[4][16 * 64];

  const int wave = threadIdx.x >> 5;
  const int lane = threadIdx.x & 31;
  const int tileBase = (blockIdx.x * 4 + wave) << 4;
  if (tileBase >= E) return;                       // wave-uniform

  _Float16* tile = sTile[wave];
  _Float16* hid  = sHid[wave];

  // stage [x_src | x_dst | e | 0-pad] as f16 rows of Kpad
  for (int r = 0; r < 16; ++r) {
    const int e = tileBase + r;
    const int s = srcIdx[e];
    const int d = dstIdx[e];
    for (int k = lane; k < Kpad; k += 32) {
      float v = 0.f;
      if (k < nin)               v = nodeSrc[(size_t)s * nin + k];
      else if (k < 2 * nin)      v = nodeSrc[(size_t)d * nin + (k - nin)];
      else if (k < 2 * nin + ein) {
        const int kk = k - 2 * nin;
        v = eIsF16 ? (float)ebuf[(size_t)e * 64 + kk]
                   : eattrF32[(size_t)e * ein + kk];
      }
      tile[r * Kpad + k] = (_Float16)v;
    }
  }

  v8f acc2[4];
  wave_mlp(tile, Kpad, hid, W1T, b1, g, be, W2T, outTiles, lane, acc2);

  const int half = lane >> 4;
  const int l16  = lane & 15;
  for (int nt = 0; nt < outTiles; ++nt) {
    const int col = (nt << 4) + l16;
    const float bb = (col < outValid) ? b2[col] : 0.f;
#pragma unroll
    for (int i = 0; i < 8; ++i) {
      const int e   = tileBase + i + (half << 3);
      const float enew = acc2[nt][i] + bb;
      const int d   = dstIdx[e];
      if (mode == 2) {
        if (col == 0) {
          edgeOutF32[e] = enew;                    // final edge output (eout=1)
          atomicAdd(&agg[d], enew);                // width-1 segment sum
        }
      } else {
        atomicAdd(&agg[(size_t)d * 64 + col], enew);   // segment_sum(e_new)
        float sto = enew;
        if (mode == 1) sto += (float)ebuf[(size_t)e * 64 + col]; // residual
        ebuf[(size_t)e * 64 + col] = (_Float16)sto;
      }
    }
  }
}

// ---------------------------------------------------------------------------
// Node MLP.  mode: 0 = store nodeFeat, 1 = residual nodeFeat += n_new,
//            2 = final (write d_out node section, width outValid)
// ---------------------------------------------------------------------------
__global__ __launch_bounds__(128) void node_mlp_kernel(
    int N, const float* nodeSrc, int nin,
    const float* __restrict__ agg, int aw, int Kpad,
    const _Float16* __restrict__ W1T, const float* __restrict__ b1,
    const float* __restrict__ g,     const float* __restrict__ be,
    const _Float16* __restrict__ W2T, const float* __restrict__ b2,
    int outTiles, int outValid, int mode,
    float* nodeFeat, float* __restrict__ outF32)
{
  __shared__ _Float16 sTile[4][16 * 192];
  __shared__ _Float16 sHid[4][16 * 64];

  const int wave = threadIdx.x >> 5;
  const int lane = threadIdx.x & 31;
  const int tileBase = (blockIdx.x * 4 + wave) << 4;
  if (tileBase >= N) return;                       // wave-uniform

  _Float16* tile = sTile[wave];
  _Float16* hid  = sHid[wave];

  for (int r = 0; r < 16; ++r) {
    const int n = tileBase + r;
    for (int k = lane; k < Kpad; k += 32) {
      float v = 0.f;
      if (k < nin)            v = nodeSrc[(size_t)n * nin + k];
      else if (k < nin + aw)  v = agg[(size_t)n * aw + (k - nin)];
      tile[r * Kpad + k] = (_Float16)v;
    }
  }

  v8f acc2[4];
  wave_mlp(tile, Kpad, hid, W1T, b1, g, be, W2T, outTiles, lane, acc2);

  const int half = lane >> 4;
  const int l16  = lane & 15;
  for (int nt = 0; nt < outTiles; ++nt) {
    const int col = (nt << 4) + l16;
    const float bb = (col < outValid) ? b2[col] : 0.f;
#pragma unroll
    for (int i = 0; i < 8; ++i) {
      const int n = tileBase + i + (half << 3);
      const float val = acc2[nt][i] + bb;
      if (mode == 2) {
        if (col < outValid) outF32[(size_t)n * outValid + col] = val;
      } else if (mode == 1) {
        nodeFeat[(size_t)n * 64 + col] += val;     // residual (row-exclusive)
      } else {
        nodeFeat[(size_t)n * 64 + col] = val;
      }
    }
  }
}

// ---------------------------------------------------------------------------
// Helpers: weight transpose/convert f32[K][C] -> f16[Cpad][Kpad], and zero fill
// ---------------------------------------------------------------------------
__global__ void transpose_pad_f16(const float* __restrict__ src,
                                  _Float16* __restrict__ dst,
                                  int K, int C, int Kpad, int Cpad)
{
  const int idx = blockIdx.x * blockDim.x + threadIdx.x;
  if (idx >= Kpad * Cpad) return;
  const int c = idx / Kpad;
  const int k = idx - c * Kpad;
  const float v = (k < K && c < C) ? src[(size_t)k * C + c] : 0.f;
  dst[idx] = (_Float16)v;
}

__global__ void zero_f32(float* __restrict__ p, int n)
{
  const int i = blockIdx.x * blockDim.x + threadIdx.x;
  if (i < n) p[i] = 0.f;
}

// ---------------------------------------------------------------------------
extern "C" void kernel_launch(void* const* d_in, const int* in_sizes, int n_in,
                              void* d_out, int out_size, void* d_ws, size_t ws_size,
                              hipStream_t stream)
{
  (void)in_sizes; (void)n_in; (void)out_size; (void)ws_size;

  const float* x     = (const float*)d_in[0];
  const int*   eidx  = (const int*)d_in[1];
  const float* eattr = (const float*)d_in[2];
  const int*   src   = eidx;
  const int*   dst   = eidx + NEDGES;

  float* nodeOut = (float*)d_out;                   // [50000][16]
  float* edgeOut = (float*)d_out + (size_t)NNODES * 16;  // [800000]

  uintptr_t w = (uintptr_t)d_ws;
  auto alloc = [&](size_t bytes) -> void* {
    void* p = (void*)w;
    w += (bytes + 255) & ~(size_t)255;
    return p;
  };
  float*    nodeFeat = (float*)alloc((size_t)NNODES * 64 * 4);
  float*    agg      = (float*)alloc((size_t)NNODES * 64 * 4);
  _Float16* ebuf     = (_Float16*)alloc((size_t)NEDGES * 64 * 2);

  // per-layer MLP dims (edge: din = 2*nin+ein ; node: din = nin + eout)
  const int eDin[4]  = {36, 192, 192, 192};
  const int eKpad[4] = {64, 192, 192, 192};
  const int eDout[4] = {64, 64, 64, 1};
  const int eTil[4]  = {4, 4, 4, 1};
  const int nDin[4]  = {80, 128, 128, 65};
  const int nKpad[4] = {96, 128, 128, 96};
  const int nDout[4] = {64, 64, 64, 16};
  const int nTil[4]  = {4, 4, 4, 1};

  _Float16 *eW1T[4], *eW2T[4], *nW1T[4], *nW2T[4];
  const float *eB1[4], *eG[4], *eBe[4], *eB2[4];
  const float *nB1[4], *nG[4], *nBe[4], *nB2[4];

  for (int l = 0; l < 4; ++l) {
    void* const* P = d_in + 3 + l * 12;
    const float* ew1 = (const float*)P[0];
    eB1[l] = (const float*)P[1]; eG[l] = (const float*)P[2]; eBe[l] = (const float*)P[3];
    const float* ew2 = (const float*)P[4];
    eB2[l] = (const float*)P[5];
    const float* nw1 = (const float*)P[6];
    nB1[l] = (const float*)P[7]; nG[l] = (const float*)P[8]; nBe[l] = (const float*)P[9];
    const float* nw2 = (const float*)P[10];
    nB2[l] = (const float*)P[11];

    eW1T[l] = (_Float16*)alloc((size_t)64 * eKpad[l] * 2);
    eW2T[l] = (_Float16*)alloc((size_t)16 * eTil[l] * 64 * 2);
    nW1T[l] = (_Float16*)alloc((size_t)64 * nKpad[l] * 2);
    nW2T[l] = (_Float16*)alloc((size_t)16 * nTil[l] * 64 * 2);

    int t;
    t = eKpad[l] * 64;
    transpose_pad_f16<<<(t + 255) / 256, 256, 0, stream>>>(ew1, eW1T[l], eDin[l], 64, eKpad[l], 64);
    t = 64 * 16 * eTil[l];
    transpose_pad_f16<<<(t + 255) / 256, 256, 0, stream>>>(ew2, eW2T[l], 64, eDout[l], 64, 16 * eTil[l]);
    t = nKpad[l] * 64;
    transpose_pad_f16<<<(t + 255) / 256, 256, 0, stream>>>(nw1, nW1T[l], nDin[l], 64, nKpad[l], 64);
    t = 64 * 16 * nTil[l];
    transpose_pad_f16<<<(t + 255) / 256, 256, 0, stream>>>(nw2, nW2T[l], 64, nDout[l], 64, 16 * nTil[l]);
  }

  const int EB = NEDGES / 64;               // 12500 blocks (exact)
  const int NB = (NNODES + 63) / 64;        // 782 blocks

  for (int l = 0; l < 4; ++l) {
    const int aggN = (l == 3) ? NNODES : NNODES * 64;
    zero_f32<<<(aggN + 255) / 256, 256, 0, stream>>>(agg, aggN);

    const int mode = (l == 0) ? 0 : ((l == 3) ? 2 : 1);
    const float* ns = (l == 0) ? x : nodeFeat;
    const int nin = (l == 0) ? 16 : 64;
    const int ein = (l == 0) ? 4 : 64;

    edge_mlp_kernel<<<EB, 128, 0, stream>>>(
        NEDGES, src, dst, ns, nin, eattr, ebuf, ein, (l > 0) ? 1 : 0,
        eKpad[l], eW1T[l], eB1[l], eG[l], eBe[l], eW2T[l], eB2[l],
        eTil[l], eDout[l], mode, edgeOut, agg);

    const int aw = (l == 3) ? 1 : 64;
    node_mlp_kernel<<<NB, 128, 0, stream>>>(
        NNODES, ns, nin, agg, aw, nKpad[l],
        nW1T[l], nB1[l], nG[l], nBe[l], nW2T[l], nB2[l],
        nTil[l], nDout[l], mode, nodeFeat, nodeOut);
  }
}
